// BiLSTM_CRF_7258494730319
// MI455X (gfx1250) — matrix operand order repaired and checked by reading, not compiled, not run on hardware
//
#include <hip/hip_runtime.h>
#include <hip/hip_bf16.h>

// ---------------------------------------------------------------------------
// BiLSTM-CRF for MI455X (gfx1250, wave32).
//  - x_proj GEMMs run on v_wmma_f32_16x16x32_f16 with f32 accumulation.
//    The per-block A tile (16 x 256 f16 = 8 KB of embeddings, shared by all
//    8 waves) is staged into LDS with GLOBAL_LOAD_ASYNC_TO_LDS_B128
//    (ASYNCcnt-tracked, s_wait_asynccnt + barrier), then read as ds_load_b128
//    fragments. B fragments stream from L2 per wave.
//  - The serial LSTM recurrence runs as two persistent 1024-thread blocks
//    (one per direction): h in LDS, W_hh pre-transposed for coalesced matvec,
//    global_prefetch_b8 on the next step's x_proj row.
//  - Viterbi: one wave, lane==tag; lane 0 backtracks.
// ---------------------------------------------------------------------------

#define T_LEN   8192
#define V_DIM   50000
#define E_DIM   256
#define H_DIM   256      // per-direction hidden
#define G_DIM   1024     // 4*H
#define HID_DIM 512
#define L_TAGS  7
#define START_TAG 5
#define STOP_TAG  6
#define NEG_VAL  (-10000.0f)

typedef __attribute__((ext_vector_type(16))) _Float16 v16h;
typedef __attribute__((ext_vector_type(8)))  _Float16 v8h;
typedef __attribute__((ext_vector_type(8)))  float    v8f;

// ---------------- prep kernels ----------------------------------------------

// Embedding gather + f32 -> f16 conversion: embF16[t*E + e]
__global__ void embed_gather_f16(const int* __restrict__ sentence,
                                 const float* __restrict__ table,
                                 _Float16* __restrict__ embF16)
{
    int idx = blockIdx.x * blockDim.x + threadIdx.x;     // over T*E
    if (idx >= T_LEN * E_DIM) return;
    int t = idx >> 8;            // E = 256
    int e = idx & (E_DIM - 1);
    embF16[idx] = (_Float16)table[(size_t)sentence[t] * E_DIM + e];
}

// W_ih_{f,b} f32 -> f16, packed [2][G_DIM*E_DIM]
__global__ void wih_to_f16(const float* __restrict__ wf,
                           const float* __restrict__ wb,
                           _Float16* __restrict__ dst)
{
    int idx = blockIdx.x * blockDim.x + threadIdx.x;     // over 2*G*E
    if (idx >= 2 * G_DIM * E_DIM) return;
    int d = idx / (G_DIM * E_DIM);
    int r = idx - d * (G_DIM * E_DIM);
    const float* src = d ? wb : wf;
    dst[idx] = (_Float16)src[r];
}

// biasSum[d][j] = b_ih[d][j] + b_hh[d][j]
__global__ void bias_sum(const float* __restrict__ bih_f, const float* __restrict__ bhh_f,
                         const float* __restrict__ bih_b, const float* __restrict__ bhh_b,
                         float* __restrict__ dst)
{
    int idx = blockIdx.x * blockDim.x + threadIdx.x;     // over 2*G
    if (idx >= 2 * G_DIM) return;
    int d = idx >> 10, j = idx & (G_DIM - 1);
    dst[idx] = d ? (bih_b[j] + bhh_b[j]) : (bih_f[j] + bhh_f[j]);
}

// whhT[d][h*G + j] = W_hh[d][j*H + h]  (f32, for coalesced recurrence matvec)
__global__ void whh_transpose(const float* __restrict__ wf,
                              const float* __restrict__ wb,
                              float* __restrict__ dst)
{
    int idx = blockIdx.x * blockDim.x + threadIdx.x;     // over 2*H*G
    if (idx >= 2 * H_DIM * G_DIM) return;
    int d = idx / (H_DIM * G_DIM);
    int r = idx - d * (H_DIM * G_DIM);
    int h = r >> 10;             // / G_DIM
    int j = r & (G_DIM - 1);
    const float* src = d ? wb : wf;
    dst[idx] = src[(size_t)j * H_DIM + h];
}

// ---------------- x_proj GEMM on WMMA ---------------------------------------
// xproj[d][t][j] = sum_e emb[t][e]*W_ih[d][j][e] + biasSum[d][j]
// grid = (T/16, 2); block = 256 (8 waves). Each wave: M=16 rows x N=128 cols.
// A tile (16x256 f16) staged once per block via async-to-LDS.
__global__ __launch_bounds__(256)
void xproj_wmma(const _Float16* __restrict__ embF16,
                const _Float16* __restrict__ wihF16,
                const float* __restrict__ biasSum,
                float* __restrict__ xproj)
{
    __shared__ __align__(16) _Float16 sA[16 * E_DIM];   // 8 KB embedding tile

    const int dir   = blockIdx.y;
    const int mtile = blockIdx.x;
    const int wave  = threadIdx.x >> 5;
    const int lane  = threadIdx.x & 31;
    const int lm    = lane & 15;     // A: row m / B,C: col n
    const int lh    = lane >> 4;     // half-wave select

    // --- async stage of A tile: 256 threads x 32 B (two b128, shared IOFFSET
    //     applies to both LDS and global addresses per ISA 08_async_tensor §4.4)
    {
        const _Float16* gsrc = embF16 + (size_t)mtile * 16 * E_DIM
                                      + (size_t)threadIdx.x * 16;
        unsigned           ldsAddr = (unsigned)(uintptr_t)(&sA[threadIdx.x * 16]);
        unsigned long long gAddr   = (unsigned long long)(uintptr_t)gsrc;
        asm volatile("global_load_async_to_lds_b128 %0, %1, off"
                     :: "v"(ldsAddr), "v"(gAddr) : "memory");
        asm volatile("global_load_async_to_lds_b128 %0, %1, off offset:16"
                     :: "v"(ldsAddr), "v"(gAddr) : "memory");
        asm volatile("s_wait_asynccnt 0x0" ::: "memory");
    }
    __syncthreads();

    const _Float16* wih  = wihF16 + (size_t)dir * G_DIM * E_DIM;
    float*          xp   = xproj  + (size_t)dir * T_LEN * G_DIM;
    const _Float16* arow = &sA[(size_t)lm * E_DIM];     // LDS-resident A row
    const int nbase = wave * 128;

    v8f acc[8] = {};
    for (int kb = 0; kb < E_DIM; kb += 32) {
        // A fragment (16x32 f16) from LDS: per-lane K = lh*16 + 0..15
        v8h alo = *(const v8h*)(arow + kb + lh * 16);       // ds_load_b128
        v8h ahi = *(const v8h*)(arow + kb + lh * 16 + 8);   // ds_load_b128
        v16h a;
        #pragma unroll
        for (int i = 0; i < 8; ++i) { a[i] = alo[i]; a[8 + i] = ahi[i]; }
        #pragma unroll
        for (int nn = 0; nn < 8; ++nn) {
            // B fragment (32x16 f16): lane col = lm, K = lh*16 + 0..15
            // B(k,n) = W_ih[n][k] -> contiguous 16 halves of W_ih row n.
            const _Float16* brow =
                wih + (size_t)(nbase + nn * 16 + lm) * E_DIM + kb + lh * 16;
            v16h b = *(const v16h*)brow;
            acc[nn] = __builtin_amdgcn_wmma_f32_16x16x32_f16(
                false, a, false, b, (short)0, acc[nn], false, false);
        }
    }
    // C/D layout: N = lm, VGPR r -> M = r + 8*lh. Bias is per-column.
    #pragma unroll
    for (int nn = 0; nn < 8; ++nn) {
        const int col = nbase + nn * 16 + lm;
        const float bs = biasSum[dir * G_DIM + col];
        #pragma unroll
        for (int r = 0; r < 8; ++r) {
            const int m = mtile * 16 + r + 8 * lh;
            xp[(size_t)m * G_DIM + col] = acc[nn][r] + bs;
        }
    }
}

// ---------------- LSTM recurrence (persistent, 1 block per direction) -------
__device__ __forceinline__ float sigm(float x) {
    return 1.0f / (1.0f + __expf(-x));
}

__global__ __launch_bounds__(1024)
void lstm_recurrence(const float* __restrict__ xproj,
                     const float* __restrict__ whhT,
                     const float* __restrict__ h0,
                     const float* __restrict__ c0,
                     float* __restrict__ lstm_out)
{
    const int dir = blockIdx.x;
    const int j   = threadIdx.x;                 // 0..1023 (gate row)
    __shared__ float h_sh[H_DIM];
    __shared__ float g_sh[G_DIM];

    const float* xp = xproj + (size_t)dir * T_LEN * G_DIM;
    const float* wt = whhT  + (size_t)dir * H_DIM * G_DIM;

    float c = 0.0f;
    if (j < H_DIM) {
        c       = c0[dir * H_DIM + j];
        h_sh[j] = h0[dir * H_DIM + j];
    }
    __syncthreads();

    for (int s = 0; s < T_LEN; ++s) {
        const int t = (dir == 0) ? s : (T_LEN - 1 - s);
        float acc = xp[(size_t)t * G_DIM + j];
        if (s + 1 < T_LEN) {
            const int t2 = (dir == 0) ? (s + 1) : (T_LEN - 2 - s);
            __builtin_prefetch(xp + (size_t)t2 * G_DIM + j, 0, 0); // global_prefetch_b8
        }
        #pragma unroll 8
        for (int h = 0; h < H_DIM; ++h)
            acc += wt[(size_t)h * G_DIM + j] * h_sh[h];   // coalesced in j
        g_sh[j] = acc;
        __syncthreads();
        if (j < H_DIM) {
            const float ig = sigm(g_sh[j]);
            const float fg = sigm(g_sh[H_DIM + j]);
            const float gg = tanhf(g_sh[2 * H_DIM + j]);
            const float og = sigm(g_sh[3 * H_DIM + j]);
            c = fg * c + ig * gg;
            const float hn = og * tanhf(c);
            h_sh[j] = hn;
            lstm_out[(size_t)t * HID_DIM + dir * H_DIM + j] = hn;
        }
        __syncthreads();
    }
}

// ---------------- output projection -----------------------------------------
__global__ void logits_proj(const float* __restrict__ lstm_out,
                            const float* __restrict__ W_out,
                            const float* __restrict__ b_out,
                            float* __restrict__ logits)
{
    int idx = blockIdx.x * blockDim.x + threadIdx.x;    // over T*L
    if (idx >= T_LEN * L_TAGS) return;
    int t = idx / L_TAGS, l = idx - t * L_TAGS;
    const float* x = lstm_out + (size_t)t * HID_DIM;
    const float* w = W_out + (size_t)l * HID_DIM;
    float acc = b_out[l];
    #pragma unroll 8
    for (int k = 0; k < HID_DIM; ++k) acc += x[k] * w[k];
    logits[idx] = acc;
}

// ---------------- Viterbi (one wave; lane == next-tag) -----------------------
__global__ void viterbi(const float* __restrict__ logits,
                        const float* __restrict__ trans,
                        int* __restrict__ bp,
                        float* __restrict__ out)
{
    const int l = threadIdx.x;
    __shared__ float fv[L_TAGS];
    float tr[L_TAGS];
    if (l < L_TAGS) {
        for (int p = 0; p < L_TAGS; ++p) tr[p] = trans[l * L_TAGS + p];
        fv[l] = (l == START_TAG) ? 0.0f : NEG_VAL;
    }
    __syncthreads();
    for (int t = 0; t < T_LEN; ++t) {
        float best = -__builtin_inff();
        int bi = 0;
        if (l < L_TAGS) {
            for (int p = 0; p < L_TAGS; ++p) {
                const float v = fv[p] + tr[p];
                if (v > best) { best = v; bi = p; }      // first-max, matches argmax
            }
            bp[t * L_TAGS + l] = bi;
            best += logits[t * L_TAGS + l];
        }
        __syncthreads();
        if (l < L_TAGS) fv[l] = best;
        __syncthreads();
    }
    if (l == 0) {
        float best = -__builtin_inff();
        int bt = 0;
        for (int p = 0; p < L_TAGS; ++p) {
            const float v = fv[p] + trans[STOP_TAG * L_TAGS + p];
            if (v > best) { best = v; bt = p; }
        }
        out[0] = best;
        int tag = bt;
        for (int t = T_LEN - 1; t >= 0; --t) {
            out[1 + t] = (float)tag;
            tag = bp[t * L_TAGS + tag];
        }
    }
}

// ---------------- launcher ---------------------------------------------------
extern "C" void kernel_launch(void* const* d_in, const int* in_sizes, int n_in,
                              void* d_out, int out_size, void* d_ws, size_t ws_size,
                              hipStream_t stream)
{
    (void)in_sizes; (void)n_in; (void)out_size; (void)ws_size;
    const int*   sentence = (const int*)  d_in[0];
    const float* embed    = (const float*)d_in[1];
    const float* W_ih_f   = (const float*)d_in[2];
    const float* W_hh_f   = (const float*)d_in[3];
    const float* b_ih_f   = (const float*)d_in[4];
    const float* b_hh_f   = (const float*)d_in[5];
    const float* W_ih_b   = (const float*)d_in[6];
    const float* W_hh_b   = (const float*)d_in[7];
    const float* b_ih_b   = (const float*)d_in[8];
    const float* b_hh_b   = (const float*)d_in[9];
    const float* h0       = (const float*)d_in[10];
    const float* c0       = (const float*)d_in[11];
    const float* W_out    = (const float*)d_in[12];
    const float* b_out    = (const float*)d_in[13];
    const float* trans    = (const float*)d_in[14];
    float* out = (float*)d_out;

    char* ws = (char*)d_ws;
    size_t off = 0;
    auto carve = [&](size_t bytes) -> char* {
        char* p = ws + off;
        off += (bytes + 255) & ~(size_t)255;
        return p;
    };
    _Float16* embF16  = (_Float16*)carve((size_t)T_LEN * E_DIM * sizeof(_Float16));
    _Float16* wihF16  = (_Float16*)carve((size_t)2 * G_DIM * E_DIM * sizeof(_Float16));
    float*    biasS   = (float*)   carve((size_t)2 * G_DIM * sizeof(float));
    float*    whhT    = (float*)   carve((size_t)2 * H_DIM * G_DIM * sizeof(float));
    float*    xproj   = (float*)   carve((size_t)2 * T_LEN * G_DIM * sizeof(float));
    float*    lstmOut = (float*)   carve((size_t)T_LEN * HID_DIM * sizeof(float));
    float*    logits  = (float*)   carve((size_t)T_LEN * L_TAGS * sizeof(float));
    int*      bp      = (int*)     carve((size_t)T_LEN * L_TAGS * sizeof(int));

    embed_gather_f16<<<(T_LEN * E_DIM + 255) / 256, 256, 0, stream>>>(sentence, embed, embF16);
    wih_to_f16<<<(2 * G_DIM * E_DIM + 255) / 256, 256, 0, stream>>>(W_ih_f, W_ih_b, wihF16);
    bias_sum<<<(2 * G_DIM + 255) / 256, 256, 0, stream>>>(b_ih_f, b_hh_f, b_ih_b, b_hh_b, biasS);
    whh_transpose<<<(2 * H_DIM * G_DIM + 255) / 256, 256, 0, stream>>>(W_hh_f, W_hh_b, whhT);

    dim3 ggrid(T_LEN / 16, 2);
    xproj_wmma<<<ggrid, 256, 0, stream>>>(embF16, wihF16, biasS, xproj);

    lstm_recurrence<<<2, 1024, 0, stream>>>(xproj, whhT, h0, c0, lstmOut);

    logits_proj<<<(T_LEN * L_TAGS + 255) / 256, 256, 0, stream>>>(lstmOut, W_out, b_out, logits);

    viterbi<<<1, 32, 0, stream>>>(logits, trans, bp, out);
}
